// MoETransformer_69002944578199
// MI455X (gfx1250) — compile-verified, matrix operand order checked by reading, not run on hardware
//
#include <hip/hip_runtime.h>
#include <hip/hip_bf16.h>
#include <math.h>

// Problem dims
#define Bb   32
#define Ss   196
#define Dd   768
#define Hh   12
#define Ee   8
#define Ff   3072
#define DHd  64
#define NTOK (Bb * Ss)   // 6272 tokens, = 49 * 128

typedef __bf16 bf16_t;
typedef __attribute__((ext_vector_type(16))) __bf16 v16bf;
typedef __attribute__((ext_vector_type(8)))  float  v8f;
typedef unsigned int u32x4 __attribute__((ext_vector_type(4)));
typedef unsigned int u32x8 __attribute__((ext_vector_type(8)));

__device__ __forceinline__ v8f wmma_bf16(v16bf a, v16bf b, v8f c) {
  // D = A(16x32 bf16) * B(32x16 bf16) + C(16x16 f32)
  return __builtin_amdgcn_wmma_f32_16x16x32_bf16(false, a, false, b, (short)0, c,
                                                 false, false);
}

// A/B bf16 fragment (16x32 / 32x16): lane (l&15) selects row/col, (l>>4)
// selects the K-half; elements [0..7] = p[0..7], [8..15] = p[16..23] where p
// points at [row][ (l>>4)*8 ] of a K-major row.
__device__ __forceinline__ v16bf load_frag16(const bf16_t* p) {
  v16bf r;
#pragma unroll
  for (int i = 0; i < 8; ++i) { r[i] = p[i]; r[i + 8] = p[i + 16]; }
  return r;
}

// ------------------------------------------------------------------ TDM
// Issue one Tensor Data Mover 2-D tile load: 128 rows x 32 bf16 (64 B/row),
// row stride K elements in global memory, padded to an 80-B LDS pitch
// (pad_interval = 16 DWORDs, pad_amount = 4 DWORDs).  D# layout per
// CDNA5 ISA.  Wave-uniform; EXEC is ignored.
__device__ __forceinline__ void tdm_load_tile(unsigned lds_off,
                                              const void* gaddr,
                                              unsigned k_elems) {
  unsigned long long ga = (unsigned long long)(size_t)gaddr;
  u32x4 g0;
  g0[0] = 1u;                                          // count=1, user mode
  g0[1] = lds_off;                                     // lds_addr (bytes)
  g0[2] = (unsigned)ga;                                // global_addr[31:0]
  g0[3] = (unsigned)((ga >> 32) & 0x01FFFFFFu) | (2u << 30);  // addr[56:32]|type=2
  u32x8 g1;
  g1[0] = (1u << 16)      // data_size = 2 bytes
        | (1u << 20)      // pad_enable
        | (3u << 22)      // pad_interval: 16 DWORDs (one 64-B tile row)
        | (3u << 25);     // pad_amount: 4 DWORDs (16 B) -> 80-B pitch
  g1[1] = (32u << 16);    // tensor_dim0 = 32  (bits 79:48)
  g1[2] = (128u << 16);   // tensor_dim1 = 128 (bits 111:80)
  g1[3] = (32u << 16);    // tile_dim0 = 32    (bits 127:112)
  g1[4] = 128u;           // tile_dim1 = 128, tile_dim2 = 0
  g1[5] = k_elems;        // tensor_dim0_stride[31:0]
  g1[6] = 0u;             // stride0[47:32], stride1[15:0]
  g1[7] = 0u;
  asm volatile("tensor_load_to_lds %0, %1" ::"s"(g0), "s"(g1) : "memory");
}

// ---------------------------------------------------------------- convert
__global__ void k_f32_to_bf16(const float* __restrict__ src,
                              bf16_t* __restrict__ dst, int n) {
  int i = blockIdx.x * blockDim.x + threadIdx.x;
  if (i < n) dst[i] = (bf16_t)src[i];
}

// src: [K][N] f32  ->  dst: [N][K] bf16 (transposed weight for GEMM B-operand)
__global__ void k_f32_to_bf16_T(const float* __restrict__ src,
                                bf16_t* __restrict__ dst, int K, int N) {
  int i = blockIdx.x * blockDim.x + threadIdx.x;
  if (i < K * N) {
    int n = i / K, kk = i - n * K;
    dst[i] = (bf16_t)src[(size_t)kk * N + n];
  }
}

// ---------------------------------------------------------------- GEMM
// C[M,N] = A[M,K](bf16, row-major) * B[K,N] (given as BmT[N][K] bf16) + bias.
// Block tile 128x128, K-step 32, 256 threads (8 waves, wave = 32 rows x 64
// cols = 2x4 WMMA sub-tiles).  Both tiles staged global->LDS by the TDM with
// ping-pong double buffering: the next K-tile's DMA overlaps this tile's
// WMMAs; TENSORcnt is in-order per wave so waiting <=2 after issuing the next
// pair guarantees the current pair has landed.
// EPI: 0 = store bf16
//      1 = store bf16 scattered to V^T [b][h][dh][s]
//      2 = store f32 (+ residual `extra` [M,N])
//      3 = exact GELU -> bf16
//      4 = out(f32) += extra[row*E + expert] * (acc + bias)
template <int EPI>
__global__ __launch_bounds__(256) void k_gemm(
    const bf16_t* __restrict__ A, const bf16_t* __restrict__ BmT,
    const float* __restrict__ bias, const float* __restrict__ extra,
    void* __restrict__ outp, int M, int N, int K, int expert) {
  // LDS (dynamic, offset 0): A0 @0, B0 @10240, A1 @20480, B1 @30720 bytes.
  extern __shared__ bf16_t smem[];

  const int bn0 = blockIdx.x * 128;
  const int bm0 = blockIdx.y * 128;
  const int tid = threadIdx.x;
  const int w    = tid >> 5;
  const int lane = tid & 31;
  const int wm = w >> 1;        // 0..3 -> 32-row strip
  const int wn = w & 1;         // 0..1 -> 64-col strip
  const int r  = lane & 15;
  const int hi = lane >> 4;

  v8f zc = {};
  v8f acc[2][4];
#pragma unroll
  for (int m = 0; m < 2; ++m)
#pragma unroll
    for (int n = 0; n < 4; ++n) acc[m][n] = zc;

  // prologue: stage first K-tile into buffer 0
  if (tid == 0) {
    tdm_load_tile(0u, A + (size_t)bm0 * K, (unsigned)K);
    tdm_load_tile(10240u, BmT + (size_t)bn0 * K, (unsigned)K);
  }

  for (int kt = 0; kt < K; kt += 32) {
    const int cur = (kt >> 5) & 1;
    if (tid == 0) {
      if (kt + 32 < K) {  // prefetch next K-tile into the other buffer
        unsigned nb = (unsigned)(cur ^ 1) * 20480u;
        tdm_load_tile(nb, A + (size_t)bm0 * K + kt + 32, (unsigned)K);
        tdm_load_tile(nb + 10240u, BmT + (size_t)bn0 * K + kt + 32,
                      (unsigned)K);
        __builtin_amdgcn_s_wait_tensorcnt(2);  // current pair complete
      } else {
        __builtin_amdgcn_s_wait_tensorcnt(0);
      }
    }
    __syncthreads();  // current tiles visible to all waves

    const bf16_t* Asp = smem + (size_t)cur * 10240;  // pitch 40 bf16 (80 B)
    const bf16_t* Bsp = Asp + 5120;

    v16bf af[2], bfr[4];
#pragma unroll
    for (int m = 0; m < 2; ++m)
      af[m] = load_frag16(&Asp[(wm * 32 + m * 16 + r) * 40 + hi * 8]);
#pragma unroll
    for (int n = 0; n < 4; ++n)
      bfr[n] = load_frag16(&Bsp[(wn * 64 + n * 16 + r) * 40 + hi * 8]);
#pragma unroll
    for (int m = 0; m < 2; ++m)
#pragma unroll
      for (int n = 0; n < 4; ++n) acc[m][n] = wmma_bf16(af[m], bfr[n], acc[m][n]);
    __syncthreads();  // all waves done reading before this buffer is reused
  }

  // Epilogue. C/D layout: VGPR v <-> row (v + 8*hi), lane&15 <-> col.
#pragma unroll
  for (int m = 0; m < 2; ++m)
#pragma unroll
    for (int n = 0; n < 4; ++n)
#pragma unroll
      for (int v = 0; v < 8; ++v) {
        int row = bm0 + wm * 32 + m * 16 + v + 8 * hi;
        int col = bn0 + wn * 64 + n * 16 + r;
        float val = acc[m][n][v] + bias[col];
        if (EPI == 0) {
          ((bf16_t*)outp)[(size_t)row * N + col] = (bf16_t)val;
        } else if (EPI == 1) {
          int bb = row / Ss, si = row - bb * Ss;
          int hh = col >> 6, dh = col & 63;
          ((bf16_t*)outp)[(((size_t)(bb * Hh + hh)) * DHd + dh) * Ss + si] =
              (bf16_t)val;
        } else if (EPI == 2) {
          ((float*)outp)[(size_t)row * N + col] =
              val + extra[(size_t)row * N + col];
        } else if (EPI == 3) {
          float g = 0.5f * val * (1.0f + erff(val * 0.70710678118f));
          ((bf16_t*)outp)[(size_t)row * N + col] = (bf16_t)g;
        } else {
          ((float*)outp)[(size_t)row * N + col] +=
              extra[(size_t)row * Ee + expert] * val;
        }
      }
}

// ---------------------------------------------------------------- attention
// One wave (32 threads) per (16-query tile, head, batch). Flash-style online
// softmax over 7 key blocks of 32.  qp/kp: [token][D] bf16; vpT: [b][h][dh][s]
// bf16; ctx: [token][D] bf16.
__global__ __launch_bounds__(32) void k_attn(const bf16_t* __restrict__ qp,
                                             const bf16_t* __restrict__ kp,
                                             const bf16_t* __restrict__ vpT,
                                             bf16_t* __restrict__ ctx) {
  __shared__ float Ssc[16][32];  // P transpose scratch

  const int qt = blockIdx.x, h = blockIdx.y, b = blockIdx.z;
  const int l = threadIdx.x;
  const int r = l & 15;
  const int hi = l >> 4;
  const int q0 = qt * 16;
  const int qrow = q0 + r;

  // Q fragments: features 0..31 and 32..63 (K-dim = DH = 64)
  v16bf aq0 = {}, aq1 = {};
  if (qrow < Ss) {
    const bf16_t* qptr = qp + ((size_t)(b * Ss + qrow)) * Dd + h * DHd;
#pragma unroll
    for (int i = 0; i < 8; ++i) {
      aq0[i]     = qptr[hi * 8 + i];
      aq0[i + 8] = qptr[16 + hi * 8 + i];
      aq1[i]     = qptr[32 + hi * 8 + i];
      aq1[i + 8] = qptr[48 + hi * 8 + i];
    }
  }

  float m[8], lsum[8];
  v8f zc = {};
  v8f o[4] = {zc, zc, zc, zc};
#pragma unroll
  for (int v = 0; v < 8; ++v) { m[v] = -3e30f; lsum[v] = 0.f; }

  for (int j = 0; j < 7; ++j) {
    const int kbase = j * 32;
    v8f sh[2];
#pragma unroll
    for (int hh = 0; hh < 2; ++hh) {
      const int kidx = kbase + hh * 16 + r;  // this lane's key column
      v16bf bk0 = {}, bk1 = {};
      if (kidx < Ss) {
        const bf16_t* kptr = kp + ((size_t)(b * Ss + kidx)) * Dd + h * DHd;
#pragma unroll
        for (int i = 0; i < 8; ++i) {
          bk0[i]     = kptr[hi * 8 + i];
          bk0[i + 8] = kptr[16 + hi * 8 + i];
          bk1[i]     = kptr[32 + hi * 8 + i];
          bk1[i + 8] = kptr[48 + hi * 8 + i];
        }
      }
      v8f s = zc;
      s = wmma_bf16(aq0, bk0, s);
      s = wmma_bf16(aq1, bk1, s);
#pragma unroll
      for (int v = 0; v < 8; ++v) {
        float sv = s[v] * 0.125f;  // 1/sqrt(64)
        s[v] = (kidx < Ss) ? sv : -3e30f;
      }
      sh[hh] = s;
    }

    // online softmax row statistics (rows live in 16-lane halves)
    float corr[8];
#pragma unroll
    for (int v = 0; v < 8; ++v) {
      float mx = fmaxf(sh[0][v], sh[1][v]);
#pragma unroll
      for (int off = 1; off < 16; off <<= 1) mx = fmaxf(mx, __shfl_xor(mx, off, 32));
      float mn = fmaxf(m[v], mx);
      corr[v] = expf(m[v] - mn);
      m[v] = mn;
      float p0 = expf(sh[0][v] - mn);
      float p1 = expf(sh[1][v] - mn);
      sh[0][v] = p0;
      sh[1][v] = p1;
      float rs = p0 + p1;
#pragma unroll
      for (int off = 1; off < 16; off <<= 1) rs += __shfl_xor(rs, off, 32);
      lsum[v] = lsum[v] * corr[v] + rs;
    }
#pragma unroll
    for (int nb = 0; nb < 4; ++nb)
#pragma unroll
      for (int v = 0; v < 8; ++v) o[nb][v] *= corr[v];

    // transpose P (16x32, C-layout) into an A-fragment via LDS
    __syncthreads();
#pragma unroll
    for (int v = 0; v < 8; ++v) {
      Ssc[v + 8 * hi][r]      = sh[0][v];
      Ssc[v + 8 * hi][16 + r] = sh[1][v];
    }
    __syncthreads();
    v16bf pa;
#pragma unroll
    for (int i = 0; i < 8; ++i) {
      pa[i]     = (bf16_t)Ssc[r][hi * 8 + i];
      pa[i + 8] = (bf16_t)Ssc[r][16 + hi * 8 + i];
    }

    // O += P @ V_block : B fragments from V^T (contiguous in key index)
#pragma unroll
    for (int nb = 0; nb < 4; ++nb) {
      const int f = nb * 16 + r;  // feature column
      const bf16_t* vptr = vpT + (((size_t)(b * Hh + h)) * DHd + f) * Ss + kbase;
      v16bf bv;
#pragma unroll
      for (int i = 0; i < 8; ++i) {
        int k0i = hi * 8 + i;
        int k1i = 16 + hi * 8 + i;
        bv[i]     = (kbase + k0i < Ss) ? vptr[k0i] : (bf16_t)0.f;
        bv[i + 8] = (kbase + k1i < Ss) ? vptr[k1i] : (bf16_t)0.f;
      }
      o[nb] = wmma_bf16(pa, bv, o[nb]);
    }
  }

  // finalize: divide by row sums, store ctx (bf16, token-major)
#pragma unroll
  for (int v = 0; v < 8; ++v) {
    float inv = (lsum[v] > 0.f) ? 1.f / lsum[v] : 0.f;
#pragma unroll
    for (int nb = 0; nb < 4; ++nb) o[nb][v] *= inv;
  }
#pragma unroll
  for (int nb = 0; nb < 4; ++nb)
#pragma unroll
    for (int v = 0; v < 8; ++v) {
      int row = q0 + v + 8 * hi;
      if (row < Ss)
        ctx[((size_t)(b * Ss + row)) * Dd + h * DHd + nb * 16 + r] =
            (bf16_t)o[nb][v];
    }
}

// ---------------------------------------------------------------- layernorm
__device__ __forceinline__ float block_sum256(float v, float* sred) {
#pragma unroll
  for (int off = 16; off > 0; off >>= 1) v += __shfl_xor(v, off, 32);
  __syncthreads();
  if ((threadIdx.x & 31) == 0) sred[threadIdx.x >> 5] = v;
  __syncthreads();
  float s = 0.f;
#pragma unroll
  for (int i = 0; i < 8; ++i) s += sred[i];
  return s;
}

// out = LN(a + b) * gamma + beta ; writes f32 and/or bf16 copies
__global__ __launch_bounds__(256) void k_ln(const float* __restrict__ a,
                                            const float* __restrict__ b,
                                            const float* __restrict__ gam,
                                            const float* __restrict__ bet,
                                            float* __restrict__ outf,
                                            bf16_t* __restrict__ outb) {
  __shared__ float sred[8];
  const int row = blockIdx.x;
  const float* ar = a + (size_t)row * Dd;
  const float* br = b ? b + (size_t)row * Dd : nullptr;
  float vals[3];
  float s = 0.f;
#pragma unroll
  for (int ii = 0; ii < 3; ++ii) {
    int i = threadIdx.x + ii * 256;
    float v = ar[i] + (br ? br[i] : 0.f);
    vals[ii] = v;
    s += v;
  }
  s = block_sum256(s, sred);
  float mu = s * (1.f / Dd);
  float qv = 0.f;
#pragma unroll
  for (int ii = 0; ii < 3; ++ii) {
    float d = vals[ii] - mu;
    qv += d * d;
  }
  qv = block_sum256(qv, sred);
  float inv = rsqrtf(qv * (1.f / Dd) + 1e-5f);
#pragma unroll
  for (int ii = 0; ii < 3; ++ii) {
    int i = threadIdx.x + ii * 256;
    float ov = (vals[ii] - mu) * inv * gam[i] + bet[i];
    if (outf) outf[(size_t)row * Dd + i] = ov;
    if (outb) outb[(size_t)row * Dd + i] = (bf16_t)ov;
  }
}

// ---------------------------------------------------------------- gating
__global__ __launch_bounds__(256) void k_gate(const float* __restrict__ x,
                                              const float* __restrict__ Wg,
                                              const float* __restrict__ bg,
                                              float* __restrict__ gates) {
  __shared__ float sred[8][Ee];
  const int t = blockIdx.x;
  float acc[Ee];
#pragma unroll
  for (int e = 0; e < Ee; ++e) acc[e] = 0.f;
  for (int d = threadIdx.x; d < Dd; d += 256) {
    float xv = x[(size_t)t * Dd + d];
#pragma unroll
    for (int e = 0; e < Ee; ++e) acc[e] += xv * Wg[d * Ee + e];
  }
#pragma unroll
  for (int e = 0; e < Ee; ++e)
#pragma unroll
    for (int off = 16; off > 0; off >>= 1) acc[e] += __shfl_xor(acc[e], off, 32);
  if ((threadIdx.x & 31) == 0)
#pragma unroll
    for (int e = 0; e < Ee; ++e) sred[threadIdx.x >> 5][e] = acc[e];
  __syncthreads();
  if (threadIdx.x == 0) {
    float lg[Ee], mx = -3e30f;
    for (int e = 0; e < Ee; ++e) {
      float s = bg[e];
      for (int w = 0; w < 8; ++w) s += sred[w][e];
      lg[e] = s;
      mx = fmaxf(mx, s);
    }
    float den = 0.f;
    for (int e = 0; e < Ee; ++e) { lg[e] = expf(lg[e] - mx); den += lg[e]; }
    float inv = 1.f / den;
    for (int e = 0; e < Ee; ++e) gates[(size_t)t * Ee + e] = lg[e] * inv;
  }
}

// ---------------------------------------------------------------- launch
extern "C" void kernel_launch(void* const* d_in, const int* in_sizes, int n_in,
                              void* d_out, int out_size, void* d_ws,
                              size_t ws_size, hipStream_t stream) {
  (void)in_sizes; (void)n_in; (void)out_size; (void)ws_size;
  const float* q    = (const float*)d_in[0];
  const float* k    = (const float*)d_in[1];
  const float* v    = (const float*)d_in[2];
  const float* Wq   = (const float*)d_in[3];
  const float* bq   = (const float*)d_in[4];
  const float* Wk   = (const float*)d_in[5];
  const float* bk   = (const float*)d_in[6];
  const float* Wv   = (const float*)d_in[7];
  const float* bv   = (const float*)d_in[8];
  const float* Wo   = (const float*)d_in[9];
  const float* bo   = (const float*)d_in[10];
  const float* ln1g = (const float*)d_in[11];
  const float* ln1b = (const float*)d_in[12];
  const float* ln2g = (const float*)d_in[13];
  const float* ln2b = (const float*)d_in[14];
  const float* Wg   = (const float*)d_in[15];
  const float* bg   = (const float*)d_in[16];
  const float* W1   = (const float*)d_in[17];
  const float* b1   = (const float*)d_in[18];
  const float* W2   = (const float*)d_in[19];
  const float* b2   = (const float*)d_in[20];

  char* wsb = (char*)d_ws;
  size_t off = 0;
  auto alloc = [&](size_t bytes) -> void* {
    off = (off + 255) & ~(size_t)255;
    void* p = wsb + off;
    off += bytes;
    return p;
  };

  const size_t ND = (size_t)NTOK * Dd;
  // Transposed bf16 weights (B-operands, [N][K])
  bf16_t* tWq = (bf16_t*)alloc((size_t)Dd * Dd * 2);
  bf16_t* tWk = (bf16_t*)alloc((size_t)Dd * Dd * 2);
  bf16_t* tWv = (bf16_t*)alloc((size_t)Dd * Dd * 2);
  bf16_t* tWo = (bf16_t*)alloc((size_t)Dd * Dd * 2);
  bf16_t* tW1 = (bf16_t*)alloc((size_t)Ee * Dd * Ff * 2);  // per expert [F][D]
  bf16_t* tW2 = (bf16_t*)alloc((size_t)Ee * Ff * Dd * 2);  // per expert [D][F]
  bf16_t* bqi = (bf16_t*)alloc(ND * 2);
  bf16_t* bki = (bf16_t*)alloc(ND * 2);
  bf16_t* bvi = (bf16_t*)alloc(ND * 2);
  bf16_t* qp  = (bf16_t*)alloc(ND * 2);
  bf16_t* kp  = (bf16_t*)alloc(ND * 2);
  bf16_t* vpT = (bf16_t*)alloc(ND * 2);
  bf16_t* ctx = (bf16_t*)alloc(ND * 2);
  float*  xres = (float*)alloc(ND * 4);
  float*  xf   = (float*)alloc(ND * 4);
  bf16_t* xb   = (bf16_t*)alloc(ND * 2);
  float*  gates = (float*)alloc((size_t)NTOK * Ee * 4);
  bf16_t* hbuf  = (bf16_t*)alloc((size_t)NTOK * Ff * 2);
  float*  moe   = (float*)alloc(ND * 4);

  auto conv = [&](const float* s, bf16_t* d, size_t n) {
    k_f32_to_bf16<<<dim3((unsigned)((n + 255) / 256)), 256, 0, stream>>>(s, d,
                                                                         (int)n);
  };
  auto convT = [&](const float* s, bf16_t* d, int K, int N) {
    size_t n = (size_t)K * N;
    k_f32_to_bf16_T<<<dim3((unsigned)((n + 255) / 256)), 256, 0, stream>>>(s, d,
                                                                           K, N);
  };
  conv(q, bqi, ND);
  conv(k, bki, ND);
  conv(v, bvi, ND);
  convT(Wq, tWq, Dd, Dd);
  convT(Wk, tWk, Dd, Dd);
  convT(Wv, tWv, Dd, Dd);
  convT(Wo, tWo, Dd, Dd);
  for (int e = 0; e < Ee; ++e) {
    convT(W1 + (size_t)e * Dd * Ff, tW1 + (size_t)e * Dd * Ff, Dd, Ff);
    convT(W2 + (size_t)e * Ff * Dd, tW2 + (size_t)e * Ff * Dd, Ff, Dd);
  }

  const unsigned SM = 40960;               // 2 x (A,B) ping-pong LDS tiles
  const dim3 gProj(Dd / 128, NTOK / 128);  // 6 x 49
  // Q/K/V projections
  k_gemm<0><<<gProj, 256, SM, stream>>>(bqi, tWq, bq, nullptr, qp, NTOK, Dd, Dd, 0);
  k_gemm<0><<<gProj, 256, SM, stream>>>(bki, tWk, bk, nullptr, kp, NTOK, Dd, Dd, 0);
  k_gemm<1><<<gProj, 256, SM, stream>>>(bvi, tWv, bv, nullptr, vpT, NTOK, Dd, Dd, 0);

  // attention: 13 query tiles x 12 heads x 32 batches, 1 wave each
  k_attn<<<dim3(13, Hh, Bb), 32, 0, stream>>>(qp, kp, vpT, ctx);

  // output projection + residual(q)
  k_gemm<2><<<gProj, 256, SM, stream>>>(ctx, tWo, bo, q, xres, NTOK, Dd, Dd, 0);

  // LN1 -> x (f32 + bf16)
  k_ln<<<NTOK, 256, 0, stream>>>(xres, nullptr, ln1g, ln1b, xf, xb);

  // gating softmax
  k_gate<<<NTOK, 256, 0, stream>>>(xf, Wg, bg, gates);

  // MoE accumulation buffer
  hipMemsetAsync(moe, 0, ND * 4, stream);

  const dim3 gUp(Ff / 128, NTOK / 128);   // 24 x 49
  const dim3 gDn(Dd / 128, NTOK / 128);   // 6 x 49
  for (int e = 0; e < Ee; ++e) {
    k_gemm<3><<<gUp, 256, SM, stream>>>(xb, tW1 + (size_t)e * Dd * Ff,
                                        b1 + (size_t)e * Ff, nullptr, hbuf, NTOK,
                                        Ff, Dd, e);
    k_gemm<4><<<gDn, 256, SM, stream>>>(hbuf, tW2 + (size_t)e * Ff * Dd,
                                        b2 + (size_t)e * Dd, gates, moe, NTOK,
                                        Dd, Ff, e);
  }

  // LN2 over (x + moe) -> output (f32)
  k_ln<<<NTOK, 256, 0, stream>>>(xf, moe, ln2g, ln2b, (float*)d_out, nullptr);
}
